// Multihead_Attention_71768903516373
// MI455X (gfx1250) — matrix operand order repaired and checked by reading, not compiled, not run on hardware
//
#include <hip/hip_runtime.h>
#include <hip/hip_bf16.h>

// ---------------------------------------------------------------------------
// Multi-head attention for MI455X (gfx1250), wave32 + WMMA bf16 + async-to-LDS.
//   B=4, S=2048, D_MODEL=1024, H=16, DK=DV=64
// Pipeline:
//   wconv_kernel    : pre-transpose+convert weights to bf16 [c][d] (one-shot)
//   proj_kernel  x3 : Qh/Kh/Vh = X @ W  (bf16 out, scale folded into Qh)
//   attn_kernel     : flash attention per (b,h); 64-key chunks; softmax
//                     denominator accumulated via WMMA ones-column trick
//   outproj_kernel  : concat @ W_O -> f32 output
// Workspace: 75,497,472 bytes.
// ---------------------------------------------------------------------------

#define B_   4
#define S_   2048
#define DM_  1024
#define H_   16
#define DV_  64

typedef __bf16 bf16_t;
typedef __attribute__((ext_vector_type(16))) __bf16 v16bf;
typedef __attribute__((ext_vector_type(8)))  float  v8f;

// D = A(16x32 bf16) * B(32x16 bf16) + C(16x16 f32)
__device__ __forceinline__ v8f wmma_bf16(v16bf a, v16bf b, v8f c) {
  return __builtin_amdgcn_wmma_f32_16x16x32_bf16(false, a, false, b,
                                                 (short)0, c, false, false);
}

// A-fragment loader (16x32, 16-bit): lane half kgrp selects contiguous K
// range. VGPR p holds K = kgrp*16+2p + {0,1}.
__device__ __forceinline__ v16bf load_a_bf16(const bf16_t* baseRow, int kgrp) {
  v16bf a;
#pragma unroll
  for (int p = 0; p < 8; ++p) {
    int off = kgrp * 16 + 2 * p;
    a[2 * p]     = baseRow[off];
    a[2 * p + 1] = baseRow[off + 1];
  }
  return a;
}

__device__ __forceinline__ v16bf load_a_f32(const float* baseRow, int kgrp) {
  v16bf a;
#pragma unroll
  for (int p = 0; p < 8; ++p) {
    int off = kgrp * 16 + 2 * p;
    float2 f = *(const float2*)(baseRow + off);
    a[2 * p]     = (bf16_t)f.x;
    a[2 * p + 1] = (bf16_t)f.y;
  }
  return a;
}

// B-fragment: lane = column, 16 contiguous bf16 along K (32B aligned).
__device__ __forceinline__ v16bf load_b_contig(const bf16_t* p) {
  return *(const v16bf*)p;
}

// Async copy of one 16-byte chunk global -> LDS (GV addressing, ASYNCcnt).
__device__ __forceinline__ void async_copy_b128(unsigned lds_byte_addr,
                                                const void* gptr) {
  asm volatile("global_load_async_to_lds_b128 %0, %1, off"
               :: "v"(lds_byte_addr),
                  "v"((unsigned long long)(size_t)gptr)
               : "memory");
}
__device__ __forceinline__ void wait_async0() {
  asm volatile("s_wait_asynccnt 0" ::: "memory");
}

// ---------------------------------------------------------------------------
// Kernel 0: weight transpose/convert.  Wt[h][c][d] = (bf16) W[h][d][c]
// R = 1024 (d). Grid: (R/32, C/64, nHead), block 256.
// ---------------------------------------------------------------------------
__global__ __launch_bounds__(256) void wconv_kernel(
    const float* __restrict__ W, bf16_t* __restrict__ Wt, int C) {
  __shared__ bf16_t lt[64 * 33];
  const int h  = blockIdx.z;
  const int r0 = blockIdx.x * 32;   // d base
  const int c0 = blockIdx.y * 64;   // c base
  const float* in = W + (size_t)h * 1024 * C;
  bf16_t* out = Wt + (size_t)h * C * 1024;

  for (int i = threadIdx.x; i < 2048; i += 256) {
    int dd = i >> 6, c = i & 63;                     // coalesced in c
    lt[c * 33 + dd] = (bf16_t)in[(size_t)(r0 + dd) * C + c0 + c];
  }
  __syncthreads();
  for (int i = threadIdx.x; i < 2048; i += 256) {
    int c = i >> 5, dd = i & 31;                     // coalesced in d
    out[(size_t)(c0 + c) * 1024 + r0 + dd] = lt[c * 33 + dd];
  }
}

// ---------------------------------------------------------------------------
// Kernel 1: Yh[b,h,s,k] = scale * sum_d X[b,s,d] * Wt[h,k,d]   (bf16 out)
// Grid: (B*S/64, H), block 128 (4 waves).
// ---------------------------------------------------------------------------
__global__ __launch_bounds__(128) void proj_kernel(
    const float* __restrict__ X,     // [B,S,DM]
    const bf16_t* __restrict__ Wt,   // [H,64,DM] pre-transposed bf16
    bf16_t* __restrict__ Y,          // [B,H,S,64]
    float scale) {
  __shared__ bf16_t wstage[64 * 32] __attribute__((aligned(32)));  // [c][d]

  const int h    = blockIdx.y;
  const int lane = threadIdx.x & 31;
  const int wave = threadIdx.x >> 5;
  const int kgrp = lane >> 4;
  const int col  = lane & 15;
  const int rowTile = blockIdx.x * 64 + wave * 16;
  const float* xrow = X + (size_t)(rowTile + col) * DM_;
  const bf16_t* wt  = Wt + (size_t)h * 64 * DM_;
  const unsigned ldsbase = (unsigned)(size_t)wstage;

  v8f acc[4] = {};

  for (int d0 = 0; d0 < DM_; d0 += 32) {
    __syncthreads();
    for (int i = threadIdx.x; i < 256; i += 128) {
      int r = i >> 2, q = i & 3;
      async_copy_b128(ldsbase + (unsigned)(r * 64 + q * 16),
                      wt + (size_t)r * DM_ + d0 + q * 8);
    }
    wait_async0();
    __syncthreads();

    v16bf a = load_a_f32(xrow + d0, kgrp);
    v16bf bfrag[4];
#pragma unroll
    for (int t = 0; t < 4; ++t)
      bfrag[t] = load_b_contig(&wstage[(t * 16 + col) * 32 + kgrp * 16]);
#pragma unroll
    for (int t = 0; t < 4; ++t)
      acc[t] = wmma_bf16(a, bfrag[t], acc[t]);
  }

  const int bIdx  = rowTile / S_;
  const int sBase = rowTile % S_;
  bf16_t* yb = Y + (((size_t)bIdx * H_ + h) * S_ + sBase) * 64;
#pragma unroll
  for (int t = 0; t < 4; ++t)
#pragma unroll
    for (int i = 0; i < 8; ++i) {
      int m = i + 8 * kgrp;  // C layout: VGPR i holds row i + 8*(lane>>4)
      yb[(size_t)m * 64 + t * 16 + col] = (bf16_t)(acc[t][i] * scale);
    }
}

// ---------------------------------------------------------------------------
// Kernel 2: flash attention per (b,h). Grid (S/128, H, B), block 256 (8 waves).
// 64-key chunks; softmax denominator via extra WMMA against all-ones B.
// ---------------------------------------------------------------------------
__global__ __launch_bounds__(256) void attn_kernel(
    const bf16_t* __restrict__ Qh,  // [B,H,S,64], pre-scaled by 1/sqrt(S)
    const bf16_t* __restrict__ Kh,
    const bf16_t* __restrict__ Vh,
    bf16_t* __restrict__ Cc) {      // [B,S,H*64] head-major concat
  __shared__ bf16_t kstage[64 * 64]     __attribute__((aligned(32)));  // [key][d]
  __shared__ bf16_t vtstage[64 * 64]    __attribute__((aligned(32)));  // [vdim][key]
  __shared__ bf16_t pstage[8][16 * 64]  __attribute__((aligned(32)));  // per-wave P

  const int b    = blockIdx.z;
  const int h    = blockIdx.y;
  const int lane = threadIdx.x & 31;
  const int wave = threadIdx.x >> 5;
  const int kgrp = lane >> 4;
  const int col  = lane & 15;
  const size_t headBase = ((size_t)b * H_ + h) * S_ * 64;
  const int qTile = blockIdx.x * 128 + wave * 16;
  const unsigned ksbase = (unsigned)(size_t)kstage;

  const bf16_t* qrow = Qh + headBase + (size_t)(qTile + col) * 64;
  v16bf aq0 = load_a_bf16(qrow, kgrp);        // d = 0..31
  v16bf aq1 = load_a_bf16(qrow + 32, kgrp);   // d = 32..63

  v16bf ones;
#pragma unroll
  for (int e = 0; e < 16; ++e) ones[e] = (bf16_t)1.0f;

  v8f acc[4] = {};
  v8f accl = {};                       // running sum(exp) per row (x16 cols)
  float rmax[8];
#pragma unroll
  for (int i = 0; i < 8; ++i) rmax[i] = -1e30f;

  for (int kb = 0; kb < S_; kb += 64) {
    const bf16_t* kc = Kh + headBase + (size_t)kb * 64;
    const bf16_t* vc = Vh + headBase + (size_t)kb * 64;
    __syncthreads();
    // K chunk: straight async copy (512 x 16B = 8KB)
    async_copy_b128(ksbase + threadIdx.x * 16u, kc + threadIdx.x * 8);
    async_copy_b128(ksbase + 4096u + threadIdx.x * 16u, kc + 2048 + threadIdx.x * 8);
    // V chunk: transposed staging through registers (dword loads)
    {
      const unsigned int* vc4 = (const unsigned int*)vc;
      for (int i = threadIdx.x; i < 2048; i += 256) {
        unsigned int u = vc4[i];
        int key = i >> 5;
        int d2  = (i & 31) * 2;
        vtstage[d2 * 64 + key]       = ((const bf16_t*)&u)[0];
        vtstage[(d2 + 1) * 64 + key] = ((const bf16_t*)&u)[1];
      }
    }
    if (kb + 64 < S_) __builtin_prefetch(vc + 4096, 0, 0);
    wait_async0();
    __syncthreads();

    // scores: four 16x16 key tiles, K=64 via two chained WMMAs each
    v8f sc[4];
#pragma unroll
    for (int tt = 0; tt < 4; ++tt) {
      v16bf bk0 = load_b_contig(&kstage[(tt * 16 + col) * 64 + kgrp * 16]);
      v16bf bk1 = load_b_contig(&kstage[(tt * 16 + col) * 64 + 32 + kgrp * 16]);
      v8f z = {};
      z = wmma_bf16(aq0, bk0, z);
      sc[tt] = wmma_bf16(aq1, bk1, z);
    }

    // online softmax: one 16-lane max reduction per row per 64 keys
#pragma unroll
    for (int i = 0; i < 8; ++i) {
      float mx = fmaxf(fmaxf(sc[0][i], sc[1][i]), fmaxf(sc[2][i], sc[3][i]));
#pragma unroll
      for (int m = 1; m < 16; m <<= 1) mx = fmaxf(mx, __shfl_xor(mx, m, 32));
      float nm   = fmaxf(rmax[i], mx);
      float corr = __expf(rmax[i] - nm);
      rmax[i] = nm;
      int mRow = i + 8 * kgrp;
#pragma unroll
      for (int tt = 0; tt < 4; ++tt) {
        float p = __expf(sc[tt][i] - nm);
        pstage[wave][mRow * 64 + tt * 16 + col] = (bf16_t)p;
      }
      acc[0][i] *= corr; acc[1][i] *= corr;
      acc[2][i] *= corr; acc[3][i] *= corr;
      accl[i]   *= corr;
    }
    // wave-private C-layout -> A-layout transpose through LDS (per-wave DS
    // ops are in-order; drain counter before reading back cross-lane).
    asm volatile("s_wait_dscnt 0" ::: "memory");

    v16bf ap0 = load_a_bf16(&pstage[wave][col * 64], kgrp);       // keys 0..31
    v16bf ap1 = load_a_bf16(&pstage[wave][col * 64 + 32], kgrp);  // keys 32..63
    v16bf bv0[4], bv1[4];
#pragma unroll
    for (int t = 0; t < 4; ++t) {
      bv0[t] = load_b_contig(&vtstage[(t * 16 + col) * 64 + kgrp * 16]);
      bv1[t] = load_b_contig(&vtstage[(t * 16 + col) * 64 + 32 + kgrp * 16]);
    }
#pragma unroll
    for (int t = 0; t < 4; ++t) {
      acc[t] = wmma_bf16(ap0, bv0[t], acc[t]);
      acc[t] = wmma_bf16(ap1, bv1[t], acc[t]);
    }
    accl = wmma_bf16(ap0, ones, accl);   // softmax denominator accumulation
    accl = wmma_bf16(ap1, ones, accl);
  }

  bf16_t* outb = Cc + ((size_t)b * S_ + qTile) * (H_ * DV_) + h * DV_;
#pragma unroll
  for (int i = 0; i < 8; ++i) {
    float inv = 1.0f / accl[i];          // row sum, replicated across columns
    int m = i + 8 * kgrp;
#pragma unroll
    for (int t = 0; t < 4; ++t)
      outb[(size_t)m * (H_ * DV_) + t * 16 + col] = (bf16_t)(acc[t][i] * inv);
  }
}

// ---------------------------------------------------------------------------
// Kernel 3: Out[r, c] = sum_d Cc[r,d] * WOt[c,d], f32 out.
// Grid (B*S/64, 1024/64), block 128 (4 waves).
// ---------------------------------------------------------------------------
__global__ __launch_bounds__(128) void outproj_kernel(
    const bf16_t* __restrict__ Cc,   // [B*S, 1024] bf16
    const bf16_t* __restrict__ WOt,  // [1024 c][1024 d] bf16 (pre-transposed)
    float* __restrict__ Out) {       // [B*S, 1024] f32
  __shared__ bf16_t wstage[64 * 32] __attribute__((aligned(32)));

  const int lane = threadIdx.x & 31;
  const int wave = threadIdx.x >> 5;
  const int kgrp = lane >> 4;
  const int col  = lane & 15;
  const int row0 = blockIdx.x * 64 + wave * 16;
  const int cb   = blockIdx.y * 64;
  const bf16_t* arow = Cc + (size_t)(row0 + col) * 1024;
  const unsigned ldsbase = (unsigned)(size_t)wstage;

  v8f acc[4] = {};
  for (int d0 = 0; d0 < 1024; d0 += 32) {
    __syncthreads();
    for (int i = threadIdx.x; i < 256; i += 128) {
      int r = i >> 2, q = i & 3;
      async_copy_b128(ldsbase + (unsigned)(r * 64 + q * 16),
                      WOt + (size_t)(cb + r) * 1024 + d0 + q * 8);
    }
    wait_async0();
    __syncthreads();

    v16bf a = load_a_bf16(arow + d0, kgrp);
    v16bf bfrag[4];
#pragma unroll
    for (int t = 0; t < 4; ++t)
      bfrag[t] = load_b_contig(&wstage[(t * 16 + col) * 32 + kgrp * 16]);
#pragma unroll
    for (int t = 0; t < 4; ++t)
      acc[t] = wmma_bf16(a, bfrag[t], acc[t]);
  }
  float* ob = Out + (size_t)row0 * 1024 + cb;
#pragma unroll
  for (int t = 0; t < 4; ++t)
#pragma unroll
    for (int i = 0; i < 8; ++i) {
      int m = i + 8 * kgrp;
      ob[(size_t)m * 1024 + t * 16 + col] = acc[t][i];
    }
}

// ---------------------------------------------------------------------------
extern "C" void kernel_launch(void* const* d_in, const int* in_sizes, int n_in,
                              void* d_out, int out_size, void* d_ws, size_t ws_size,
                              hipStream_t stream) {
  (void)in_sizes; (void)n_in; (void)out_size; (void)ws_size;
  const float* Q  = (const float*)d_in[0];
  const float* K  = (const float*)d_in[1];
  const float* V  = (const float*)d_in[2];
  const float* WQ = (const float*)d_in[3];
  const float* WK = (const float*)d_in[4];
  const float* WV = (const float*)d_in[5];
  const float* WO = (const float*)d_in[6];
  float* Out = (float*)d_out;

  const size_t headElems = (size_t)B_ * H_ * S_ * DV_;  // 8,388,608
  const size_t wElems    = (size_t)H_ * DV_ * DM_;      // 1,048,576
  bf16_t* Qh  = (bf16_t*)d_ws;
  bf16_t* Kh  = Qh + headElems;
  bf16_t* Vh  = Kh + headElems;
  bf16_t* Cc  = Vh + headElems;        // [B,S,H*64]
  bf16_t* WQt = Cc + headElems;
  bf16_t* WKt = WQt + wElems;
  bf16_t* WVt = WKt + wElems;
  bf16_t* WOt = WVt + wElems;          // [1024,1024]

  const float scale = 0.02209708691207961f;  // 1/sqrt(S) — reference uses seq len!

  wconv_kernel<<<dim3(32, 1, H_), dim3(256), 0, stream>>>(WQ, WQt, 64);
  wconv_kernel<<<dim3(32, 1, H_), dim3(256), 0, stream>>>(WK, WKt, 64);
  wconv_kernel<<<dim3(32, 1, H_), dim3(256), 0, stream>>>(WV, WVt, 64);
  wconv_kernel<<<dim3(32, 16, 1), dim3(256), 0, stream>>>(WO, WOt, 1024);

  dim3 gProj((B_ * S_) / 64, H_);
  proj_kernel<<<gProj, dim3(128), 0, stream>>>(Q, WQt, Qh, scale);
  proj_kernel<<<gProj, dim3(128), 0, stream>>>(K, WKt, Kh, 1.0f);
  proj_kernel<<<gProj, dim3(128), 0, stream>>>(V, WVt, Vh, 1.0f);

  dim3 gAttn(S_ / 128, H_, B_);
  attn_kernel<<<gAttn, dim3(256), 0, stream>>>(Qh, Kh, Vh, Cc);

  dim3 gOut((B_ * S_) / 64, 1024 / 64);
  outproj_kernel<<<gOut, dim3(128), 0, stream>>>(Cc, WOt, Out);
}
